// DISTAtteNCionEDual_52810917871763
// MI455X (gfx1250) — compile-verified, matrix operand order checked by reading, not run on hardware
//
#include <hip/hip_runtime.h>

typedef __attribute__((ext_vector_type(16))) _Float16 v16h;
typedef __attribute__((ext_vector_type(8)))  _Float16 v8h;
typedef __attribute__((ext_vector_type(4)))  _Float16 v4h;
typedef __attribute__((ext_vector_type(8)))  float    v8f;

#define S_DIM 224
#define SSQ   50176      // 224*224
#define CTRI  128

static __device__ __forceinline__ float sigmoidf_(float x){ return 1.f/(1.f+__expf(-x)); }
static __device__ __forceinline__ float wred_max(float v){
#pragma unroll
  for (int o=16;o>0;o>>=1) v = fmaxf(v, __shfl_xor(v,o,32));
  return v;
}
static __device__ __forceinline__ float wred_sum(float v){
#pragma unroll
  for (int o=16;o>0;o>>=1) v += __shfl_xor(v,o,32);
  return v;
}
static __device__ __forceinline__ unsigned rotl32(unsigned x, int r){ return (x<<r)|(x>>(32-r)); }
static __device__ __forceinline__ v4h cvt4(float4 v){
  v4h r; r[0]=(_Float16)v.x; r[1]=(_Float16)v.y; r[2]=(_Float16)v.z; r[3]=(_Float16)v.w; return r;
}

// ---------------------------------------------------------------------------
// Generic fused GEMM: Cout[M,N] = post( (A[M,K] @ W[K,N]) * scale + bias )
// post: act(0=none,1=sigmoid,2=relu), optional *mul[M,N], *mrow[M], +addp[M,N].
// f16 WMMA, f32 accum. Tile 128x64, BK=32, 8 waves; wave -> 32x32 (4 accums,
// 4 WMMA per K-step, 8 ds_load_b128 -> 2 loads/WMMA). W staged TRANSPOSED
// (Bt[n][k]) so all fragments are contiguous 16B runs. FASTK=false only for
// the K=33 resize GEMM (guarded scalar staging kept out of the hot kernel).
// ---------------------------------------------------------------------------
template<bool FASTK>
__global__ __launch_bounds__(256)
void gemm_kernel(const float* __restrict__ A, const float* __restrict__ W,
                 const float* __restrict__ bias, float* __restrict__ Cout,
                 int M, int K, int N, float scale, int act,
                 const float* __restrict__ mul, const float* __restrict__ mrow,
                 const float* __restrict__ addp)
{
  __shared__ __align__(16) _Float16 As[128][40];
  __shared__ __align__(16) _Float16 Bt[64][40];
  const int tid  = threadIdx.x;
  const int lane = tid & 31;
  const int wid  = tid >> 5;
  const int bm = blockIdx.y << 7;
  const int bn = blockIdx.x << 6;
  const int wm = (wid >> 1) << 5;     // 0,32,64,96
  const int wn = (wid & 1) << 5;      // 0,32
  const int l15 = lane & 15;
  const int kb  = (lane >> 4) << 3;   // 0 or 8
  v8f acc[2][2] = {};
  const int nk = (K + 31) >> 5;
  for (int kt = 0; kt < nk; ++kt) {
    const int k0 = kt << 5;
    __syncthreads();
    if (FASTK) {
      // A tile 128x32: 1024 float4 loads -> packed v4h stores (ds_store_b64)
#pragma unroll
      for (int u = 0; u < 4; ++u) {
        int q = (u << 8) + tid;          // 0..1023
        int r = q >> 3, cq = (q & 7) << 2;
        float4 v = *(const float4*)(A + (size_t)(bm + r) * K + k0 + cq);
        *(v4h*)&As[r][cq] = cvt4(v);
      }
      // W tile 32x64 staged transposed: Bt[n][k]
#pragma unroll
      for (int u = 0; u < 2; ++u) {
        int q = (u << 8) + tid;          // 0..511
        int n = q & 63, kq = (q >> 6) << 2;
        v4h pk;
#pragma unroll
        for (int i = 0; i < 4; ++i)
          pk[i] = (_Float16)W[(size_t)(k0 + kq + i) * N + bn + n];
        *(v4h*)&Bt[n][kq] = pk;
      }
      if (kt + 1 < nk) {                 // prefetch next K-tile
        int r = tid & 127;
        __builtin_prefetch(A + (size_t)(bm + r) * K + k0 + 32, 0, 1);
        if (tid < 32) __builtin_prefetch(W + (size_t)(k0 + 32 + tid) * N + bn, 0, 1);
      }
    } else {                             // guarded path (resize, K=33)
#pragma unroll
      for (int u = 0; u < 16; ++u) {
        int e = (u << 8) + tid;          // 0..4095
        int r = e >> 5, cc = e & 31;
        int kk = k0 + cc;
        float v = (kk < K) ? A[(size_t)(bm + r) * K + kk] : 0.f;
        As[r][cc] = (_Float16)v;
      }
#pragma unroll
      for (int u = 0; u < 2; ++u) {
        int e = (u << 8) + tid;          // 0..511
        int n = e & 63, kq = e >> 6;     // kq 0..7
#pragma unroll
        for (int p = 0; p < 4; ++p) {
          int kk = k0 + kq + (p << 3);
          float v = (kk < K) ? W[(size_t)kk * N + bn + n] : 0.f;
          Bt[n][kq + (p << 3)] = (_Float16)v;
        }
      }
    }
    __syncthreads();
    v16h a0f, a1f, b0f, b1f;
    ((v8h*)&a0f)[0] = *(const v8h*)&As[wm + l15][kb];
    ((v8h*)&a0f)[1] = *(const v8h*)&As[wm + l15][16 + kb];
    ((v8h*)&a1f)[0] = *(const v8h*)&As[wm + 16 + l15][kb];
    ((v8h*)&a1f)[1] = *(const v8h*)&As[wm + 16 + l15][16 + kb];
    ((v8h*)&b0f)[0] = *(const v8h*)&Bt[wn + l15][kb];
    ((v8h*)&b0f)[1] = *(const v8h*)&Bt[wn + l15][16 + kb];
    ((v8h*)&b1f)[0] = *(const v8h*)&Bt[wn + 16 + l15][kb];
    ((v8h*)&b1f)[1] = *(const v8h*)&Bt[wn + 16 + l15][16 + kb];
    acc[0][0] = __builtin_amdgcn_wmma_f32_16x16x32_f16(false, a0f, false, b0f, (short)0, acc[0][0], false, false);
    acc[0][1] = __builtin_amdgcn_wmma_f32_16x16x32_f16(false, a0f, false, b1f, (short)0, acc[0][1], false, false);
    acc[1][0] = __builtin_amdgcn_wmma_f32_16x16x32_f16(false, a1f, false, b0f, (short)0, acc[1][0], false, false);
    acc[1][1] = __builtin_amdgcn_wmma_f32_16x16x32_f16(false, a1f, false, b1f, (short)0, acc[1][1], false, false);
  }
  const int rowoff = (lane >> 4) << 3;
#pragma unroll
  for (int sub = 0; sub < 2; ++sub) {
#pragma unroll
    for (int r = 0; r < 8; ++r) {
      int m = bm + wm + (sub << 4) + r + rowoff;
#pragma unroll
      for (int half = 0; half < 2; ++half) {
        int n = bn + wn + (half << 4) + l15;
        float v = acc[sub][half][r] * scale;
        if (bias) v += bias[n];
        if (act == 1) v = sigmoidf_(v);
        else if (act == 2) v = fmaxf(v, 0.f);
        size_t gi = (size_t)m * N + n;
        if (mul)  v *= mul[gi];
        if (mrow) v *= mrow[m];
        if (addp) v += addp[gi];
        Cout[gi] = v;
      }
    }
  }
}

// ---------------------------------------------------------------------------
// LayerNorm over last dim C (template: 256 or 128). One wave per row.
// ---------------------------------------------------------------------------
template<int C>
__global__ __launch_bounds__(256)
void ln_kernel(const float* __restrict__ x, const float* __restrict__ g,
               const float* __restrict__ b, float* __restrict__ y)
{
  const int lane = threadIdx.x & 31;
  const int wid  = threadIdx.x >> 5;
  const int row  = (blockIdx.x << 3) + wid;
  const float* xr = x + (size_t)row * C;
  float* yr = y + (size_t)row * C;
  constexpr int per = C >> 5;
  float vals[per];
  float s = 0.f, ss = 0.f;
#pragma unroll
  for (int i = 0; i < per; ++i) {
    vals[i] = xr[(i << 5) + lane];
    s += vals[i]; ss += vals[i] * vals[i];
  }
  s = wred_sum(s); ss = wred_sum(ss);
  float mean = s / C;
  float var  = ss / C - mean * mean;
  float rstd = rsqrtf(var + 1e-5f);
#pragma unroll
  for (int i = 0; i < per; ++i) {
    int c = (i << 5) + lane;
    yr[c] = (vals[i] - mean) * rstd * g[c] + b[c];
  }
}

// ---------------------------------------------------------------------------
// Triangle multiplication einsum (per-channel batched GEMM, CT=128)
// mode 0 ('in') : U[i,j,c] = sum_k P[i,k,c] * Q[j,k,c]
// mode 1 ('out'): U[i,j,c] = sum_k P[k,j,c] * Q[k,i,c]
// block: 16x16 (i,j) tile x 8 channels; float4 LDS traffic.
// ---------------------------------------------------------------------------
__global__ __launch_bounds__(256)
void trimul_kernel(const float* __restrict__ P, const float* __restrict__ Q,
                   float* __restrict__ U, int mode)
{
  __shared__ __align__(16) float Ps[16][16][8];
  __shared__ __align__(16) float Qs[16][16][8];
  const int i0 = blockIdx.x << 4, j0 = blockIdx.y << 4, c0 = blockIdx.z << 3;
  const int tid = threadIdx.x;
  const int ti = tid & 15, tj = tid >> 4;
  float4 a0 = {0,0,0,0}, a1 = {0,0,0,0};
  for (int kt = 0; kt < 14; ++kt) {
    const int k0 = kt << 4;
    __syncthreads();
#pragma unroll
    for (int u = 0; u < 2; ++u) {
      int q = (u << 8) + tid;            // 0..511 float4 tasks
      int a = q >> 5, rem = q & 31, bb = rem >> 1, hf = (rem & 1) << 2;
      size_t pi, qi;
      if (mode == 0) {
        pi = ((size_t)(i0 + a) * S_DIM + (k0 + bb)) * CTRI + c0 + hf;
        qi = ((size_t)(j0 + a) * S_DIM + (k0 + bb)) * CTRI + c0 + hf;
      } else {
        pi = ((size_t)(k0 + bb) * S_DIM + (j0 + a)) * CTRI + c0 + hf;
        qi = ((size_t)(k0 + bb) * S_DIM + (i0 + a)) * CTRI + c0 + hf;
      }
      *(float4*)&Ps[a][bb][hf] = *(const float4*)(P + pi);
      *(float4*)&Qs[a][bb][hf] = *(const float4*)(Q + qi);
    }
    __syncthreads();
    const int pi0 = mode ? tj : ti;
    const int qi0 = mode ? ti : tj;
#pragma unroll
    for (int kk = 0; kk < 16; ++kk) {
      float4 p0 = *(const float4*)&Ps[pi0][kk][0];
      float4 p1 = *(const float4*)&Ps[pi0][kk][4];
      float4 q0 = *(const float4*)&Qs[qi0][kk][0];
      float4 q1 = *(const float4*)&Qs[qi0][kk][4];
      a0.x = fmaf(p0.x, q0.x, a0.x); a0.y = fmaf(p0.y, q0.y, a0.y);
      a0.z = fmaf(p0.z, q0.z, a0.z); a0.w = fmaf(p0.w, q0.w, a0.w);
      a1.x = fmaf(p1.x, q1.x, a1.x); a1.y = fmaf(p1.y, q1.y, a1.y);
      a1.z = fmaf(p1.z, q1.z, a1.z); a1.w = fmaf(p1.w, q1.w, a1.w);
    }
  }
  float* ob = U + ((size_t)(i0 + ti) * S_DIM + (j0 + tj)) * CTRI + c0;
  *(float4*)ob = a0;
  *(float4*)(ob + 4) = a1;
}

// ---------------------------------------------------------------------------
// Triangle attention, one block per (s, h, 32-wide a-tile).
// logits[a,b] = sum_c K[s,a,h,c]*Q[s,b,h,c] + pb[a,b,h] + 1e9*(mask[a,b]-1)
// softmax over b; O[a,c] = sum_b P[a,b]*V[s,b,h,c]; sigmoid gate.
// Phase 1: 64-wide b tiles, 8 waves each own a 16x16 logits tile (2 WMMA).
// Phase 3: 64-deep k tiles, Vt staged transposed, Pb zero-padded to 256.
// ---------------------------------------------------------------------------
__global__ __launch_bounds__(256)
void attn_kernel(const float* __restrict__ Km, const float* __restrict__ Qm,
                 const float* __restrict__ Vm, const float* __restrict__ PBp,
                 const float* __restrict__ maskp, const float* __restrict__ FUp,
                 float* __restrict__ OUTp)
{
  __shared__ __align__(16) _Float16 Ka[32][72];
  __shared__ __align__(16) _Float16 Tb[64][72];   // Q tiles (ph1) / V^T tiles (ph3)
  __shared__ __align__(16) float    Lg[32][228];
  __shared__ __align__(16) _Float16 Pb[32][264];  // cols 224..255 zeroed
  const int tid = threadIdx.x;
  const int lane = tid & 31, wid = tid >> 5;
  const int a0 = blockIdx.x << 5;
  const int h  = blockIdx.y;
  const int sI = blockIdx.z;
  const size_t rb = (size_t)sI * S_DIM;
  const int l15 = lane & 15;
  const int kb  = (lane >> 4) << 3;
  const int rowoff = (lane >> 4) << 3;
  // stage K rows for this a-tile (32x64)
#pragma unroll
  for (int u = 0; u < 2; ++u) {
    int q = (u << 8) + tid;
    int r = q >> 4, cq = (q & 15) << 2;
    float4 v = *(const float4*)(Km + ((rb + a0 + r) << 8) + (h << 6) + cq);
    *(v4h*)&Ka[r][cq] = cvt4(v);
  }
  // ---- phase 1: logits, 4 x 64-wide b tiles, all 8 waves ----
  for (int bt = 0; bt < 4; ++bt) {
    const int b0 = bt << 6;
    __syncthreads();
#pragma unroll
    for (int u = 0; u < 4; ++u) {
      int q = (u << 8) + tid;            // 0..1023
      int r = q >> 4, cq = (q & 15) << 2;
      int gr = b0 + r; if (gr > S_DIM - 1) gr = S_DIM - 1;   // clamp tail
      float4 v = *(const float4*)(Qm + ((rb + gr) << 8) + (h << 6) + cq);
      *(v4h*)&Tb[r][cq] = cvt4(v);
    }
    __syncthreads();
    const int ar = (wid & 1) << 4;
    const int br = (wid >> 1) << 4;      // 0,16,32,48
    v8f acc = {};
    v16h af, bf;
    ((v8h*)&af)[0] = *(const v8h*)&Ka[ar + l15][kb];
    ((v8h*)&af)[1] = *(const v8h*)&Ka[ar + l15][16 + kb];
    ((v8h*)&bf)[0] = *(const v8h*)&Tb[br + l15][kb];
    ((v8h*)&bf)[1] = *(const v8h*)&Tb[br + l15][16 + kb];
    acc = __builtin_amdgcn_wmma_f32_16x16x32_f16(false, af, false, bf, (short)0, acc, false, false);
    ((v8h*)&af)[0] = *(const v8h*)&Ka[ar + l15][32 + kb];
    ((v8h*)&af)[1] = *(const v8h*)&Ka[ar + l15][48 + kb];
    ((v8h*)&bf)[0] = *(const v8h*)&Tb[br + l15][32 + kb];
    ((v8h*)&bf)[1] = *(const v8h*)&Tb[br + l15][48 + kb];
    acc = __builtin_amdgcn_wmma_f32_16x16x32_f16(false, af, false, bf, (short)0, acc, false, false);
#pragma unroll
    for (int r = 0; r < 8; ++r) {
      int al = ar + r + rowoff;
      int gb = b0 + br + l15;
      if (gb < S_DIM) {
        size_t pix = (size_t)(a0 + al) * S_DIM + gb;
        Lg[al][gb] = acc[r] + PBp[(pix << 2) + h] + 1e9f * (maskp[pix] - 1.f);
      }
    }
  }
  __syncthreads();
  // ---- phase 2: softmax over 224 cols, 4 rows per wave; zero-pad Pb ----
#pragma unroll
  for (int q = 0; q < 4; ++q) {
    int row = (wid << 2) + q;
    float mx = -1e30f;
#pragma unroll
    for (int t = 0; t < 7; ++t) mx = fmaxf(mx, Lg[row][(t << 5) + lane]);
    mx = wred_max(mx);
    float sm = 0.f;
#pragma unroll
    for (int t = 0; t < 7; ++t) {
      float ev = __expf(Lg[row][(t << 5) + lane] - mx);
      Lg[row][(t << 5) + lane] = ev;
      sm += ev;
    }
    sm = wred_sum(sm);
    float inv = 1.f / sm;
#pragma unroll
    for (int t = 0; t < 7; ++t)
      Pb[row][(t << 5) + lane] = (_Float16)(Lg[row][(t << 5) + lane] * inv);
    Pb[row][224 + lane] = (_Float16)0.f;
  }
  // ---- phase 3: O = P @ V, 4 x 64-deep k tiles, 2 WMMA each ----
  const int ar3 = (wid >> 2) << 4;
  const int cc3 = (wid & 3) << 4;
  v8f oacc = {};
  for (int bt = 0; bt < 4; ++bt) {
    const int b0 = bt << 6;
    __syncthreads();
    // stage V tile transposed: Tb[c][b_local], 64x64
#pragma unroll
    for (int u = 0; u < 4; ++u) {
      int q = (u << 8) + tid;            // 0..1023
      int c = q & 63, rq = (q >> 6) << 2;
      v4h pk;
#pragma unroll
      for (int i = 0; i < 4; ++i) {
        int gr = b0 + rq + i; if (gr > S_DIM - 1) gr = S_DIM - 1;  // clamp; P=0 there
        pk[i] = (_Float16)Vm[((rb + gr) << 8) + (h << 6) + c];
      }
      *(v4h*)&Tb[c][rq] = pk;
    }
    __syncthreads();
    v16h af, bf;
    ((v8h*)&af)[0] = *(const v8h*)&Pb[ar3 + l15][b0 + kb];
    ((v8h*)&af)[1] = *(const v8h*)&Pb[ar3 + l15][b0 + 16 + kb];
    ((v8h*)&bf)[0] = *(const v8h*)&Tb[cc3 + l15][kb];
    ((v8h*)&bf)[1] = *(const v8h*)&Tb[cc3 + l15][16 + kb];
    oacc = __builtin_amdgcn_wmma_f32_16x16x32_f16(false, af, false, bf, (short)0, oacc, false, false);
    ((v8h*)&af)[0] = *(const v8h*)&Pb[ar3 + l15][b0 + 32 + kb];
    ((v8h*)&af)[1] = *(const v8h*)&Pb[ar3 + l15][b0 + 48 + kb];
    ((v8h*)&bf)[0] = *(const v8h*)&Tb[cc3 + l15][32 + kb];
    ((v8h*)&bf)[1] = *(const v8h*)&Tb[cc3 + l15][48 + kb];
    oacc = __builtin_amdgcn_wmma_f32_16x16x32_f16(false, af, false, bf, (short)0, oacc, false, false);
  }
#pragma unroll
  for (int r = 0; r < 8; ++r) {
    int al = ar3 + r + rowoff;
    size_t gi = ((rb + a0 + al) << 8) + (h << 6) + cc3 + l15;
    OUTp[gi] = oacc[r] * FUp[gi];
  }
}

// pair-bias projection: PB[row,h] = XL[row,:] . wb[:,h]   (N=4, tiny)
__global__ __launch_bounds__(256)
void pb_kernel(const float* __restrict__ XLp, const float* __restrict__ wb,
               float* __restrict__ PBo)
{
  int g = blockIdx.x * 256 + threadIdx.x;
  if (g >= SSQ * 4) return;
  int row = g >> 2, h = g & 3;
  const float* xr = XLp + ((size_t)row << 8);
  float a = 0.f;
  for (int c = 0; c < 256; ++c) a += xr[c] * wb[(c << 2) + h];
  PBo[g] = a;
}

// dst[(j,i),c] (+)= src[(i,j),c] * mask_at_dst_row
__global__ __launch_bounds__(256)
void transpose_kernel(const float* __restrict__ src, float* __restrict__ dst,
                      int C, int accum, const float* __restrict__ mrow)
{
  int p = blockIdx.x * 256 + threadIdx.x;
  if (p >= SSQ) return;
  int i = p / S_DIM, j = p - i * S_DIM;
  const float* sp = src + (size_t)p * C;
  size_t drow = (size_t)j * S_DIM + i;
  float* dp = dst + drow * C;
  float mk = mrow ? mrow[drow] : 1.f;
  for (int c = 0; c < C; ++c) {
    float v = sp[c] * mk;
    if (accum) dp[c] += v; else dp[c] = v;
  }
}

// X = X * mask_row + PR0    (close of _pair_update + outer residual)
__global__ __launch_bounds__(256)
void ewmadd_kernel(float* __restrict__ Xp, const float* __restrict__ PR0p,
                   const float* __restrict__ mrow)
{
  size_t g = (size_t)blockIdx.x * 256 + threadIdx.x;
  int row = (int)(g >> 8);
  Xp[g] = Xp[g] * mrow[row] + PR0p[g];
}

// ysum[s,d] = sum_j TEST[s,j,d]
__global__ __launch_bounds__(256)
void ysum_kernel(const float* __restrict__ T, float* __restrict__ YS)
{
  int sI = blockIdx.x, dI = threadIdx.x;
  float a = 0.f;
  for (int j = 0; j < S_DIM; ++j) a += T[(((size_t)sI * S_DIM + j) << 8) + dI];
  YS[(sI << 8) + dI] = a;
}

// per-s MLP head 256->64->32->4, log-softmax, CE against pair_rep diag chans 1..4
__global__ __launch_bounds__(256)
void head_kernel(const float* __restrict__ YS,
                 const float* __restrict__ m1b, const float* __restrict__ m1w,
                 const float* __restrict__ m2b, const float* __restrict__ m2w,
                 const float* __restrict__ m3b, const float* __restrict__ m3w,
                 const float* __restrict__ prep, float* __restrict__ CEo)
{
  __shared__ float h1[64], h2[32], h3[4];
  int sI = blockIdx.x, t = threadIdx.x;
  if (t < 64) {
    float a = m1b[t];
    for (int c = 0; c < 256; ++c) a += YS[(sI << 8) + c] * m1w[(c << 6) + t];
    h1[t] = sigmoidf_(a);
  }
  __syncthreads();
  if (t < 32) {
    float a = m2b[t];
    for (int c = 0; c < 64; ++c) a += h1[c] * m2w[(c << 5) + t];
    h2[t] = sigmoidf_(a);
  }
  __syncthreads();
  if (t < 4) {
    float a = m3b[t];
    for (int c = 0; c < 32; ++c) a += h2[c] * m3w[(c << 2) + t];
    h3[t] = fmaxf(a, 0.f);
  }
  __syncthreads();
  if (t == 0) {
    float mx = fmaxf(fmaxf(h3[0], h3[1]), fmaxf(h3[2], h3[3]));
    float sm = 0.f;
    for (int i = 0; i < 4; ++i) sm += __expf(h3[i] - mx);
    float lse = mx + __logf(sm);
    float ce = 0.f;
    for (int i = 0; i < 4; ++i) {
      float yv = prep[((size_t)sI * S_DIM + sI) * 33 + 1 + i];
      ce -= yv * (h3[i] - lse);
    }
    CEo[sI] = ce;
  }
}

// out[r] = relu(X[r,:].ow + ob) * mask[r], one wave per row
__global__ __launch_bounds__(256)
void outhead_kernel(const float* __restrict__ Xp, const float* __restrict__ ow,
                    const float* __restrict__ ob, const float* __restrict__ mrow,
                    float* __restrict__ outp)
{
  int lane = threadIdx.x & 31, wid = threadIdx.x >> 5;
  int row = (blockIdx.x << 3) + wid;
  const float* xr = Xp + ((size_t)row << 8);
  float a = 0.f;
#pragma unroll
  for (int i = 0; i < 8; ++i) a += xr[(i << 5) + lane] * ow[(i << 5) + lane];
  a = wred_sum(a);
  if (lane == 0) outp[row] = fmaxf(a + ob[0], 0.f) * mrow[row];
}

// l1 = sum(ce*y_drop)/max(sum(y_drop),1); u from Threefry-2x32-20, key {0,42}
__global__ __launch_bounds__(256)
void loss_kernel(const float* __restrict__ CE, const float* __restrict__ maskp,
                 float* __restrict__ l1out)
{
  __shared__ float ub[224];
  __shared__ float rn[256], rd[256];
  int t = threadIdx.x;
  if (t < 112) {
    unsigned k0 = 0u, k1 = 42u;
    unsigned ks2 = 0x1BD11BDAu ^ k0 ^ k1;
    unsigned x0 = (unsigned)t, x1 = (unsigned)(t + 112);
    x0 += k0; x1 += k1;
#define TFR(r) { x0 += x1; x1 = rotl32(x1,(r)); x1 ^= x0; }
    TFR(13) TFR(15) TFR(26) TFR(6)  x0 += k1;  x1 += ks2 + 1u;
    TFR(17) TFR(29) TFR(16) TFR(24) x0 += ks2; x1 += k0  + 2u;
    TFR(13) TFR(15) TFR(26) TFR(6)  x0 += k0;  x1 += k1  + 3u;
    TFR(17) TFR(29) TFR(16) TFR(24) x0 += k1;  x1 += ks2 + 4u;
    TFR(13) TFR(15) TFR(26) TFR(6)  x0 += ks2; x1 += k0  + 5u;
#undef TFR
    ub[t]       = __uint_as_float((x0 >> 9) | 0x3f800000u) - 1.0f;
    ub[t + 112] = __uint_as_float((x1 >> 9) | 0x3f800000u) - 1.0f;
  }
  __syncthreads();
  float num = 0.f, den = 0.f;
  if (t < 224) {
    float ym = maskp[t * S_DIM + t];
    float ydo = (ub[t] > 0.1f) ? ym : 0.f;
    float ydr = ((ydo > 0.f) ? 0.f : 1.f) * ym;
    num = CE[t] * ydr;
    den = ydr;
  }
  rn[t] = num; rd[t] = den;
  __syncthreads();
  for (int s = 128; s > 0; s >>= 1) {
    if (t < s) { rn[t] += rn[t + s]; rd[t] += rd[t + s]; }
    __syncthreads();
  }
  if (t == 0) l1out[0] = rn[0] / fmaxf(rd[0], 1.0f);
}

// ---------------------------------------------------------------------------
// Host orchestration. Param leaves follow jax tree-flatten order (sorted keys).
// ---------------------------------------------------------------------------
extern "C" void kernel_launch(void* const* d_in, const int* in_sizes, int n_in,
                              void* d_out, int out_size, void* d_ws, size_t ws_size,
                              hipStream_t stream)
{
  (void)in_sizes; (void)n_in; (void)out_size; (void)ws_size;
  const float* prep = (const float*)d_in[0];
  const float* mask = (const float*)d_in[1];
#define PW(i) ((const float*)d_in[(i)])
  float* ws = (float*)d_ws;
  const size_t BIG = (size_t)SSQ * 256;
  float* PR0 = ws;            // input to pair_update (outer residual)
  float* X   = PR0 + BIG;     // working pair rep
  float* XL  = X   + BIG;     // layernorm output
  float* B1  = XL  + BIG;
  float* B2  = B1  + BIG;
  float* B3  = B2  + BIG;
  float* B4  = B3  + BIG;
  float* B5  = B4  + BIG;
  float* PBb = B5  + BIG;
  float* MT  = PBb + (size_t)SSQ * 4;
  float* YS  = MT  + SSQ;
  float* CE  = YS  + 224 * 256;
  float* TBUF = B1;           // transition hidden (SSQ x 1024) aliases B1..B4

  dim3 blk(256, 1, 1);
  auto GEMM = [&](const float* A, const float* W, const float* bias, float* Co,
                  int M, int K, int N, float scale, int act,
                  const float* mul, const float* mrow, const float* addp){
    dim3 g(N / 64, M / 128, 1);
    if ((K & 31) == 0)
      gemm_kernel<true><<<g, blk, 0, stream>>>(A, W, bias, Co, M, K, N, scale, act, mul, mrow, addp);
    else
      gemm_kernel<false><<<g, blk, 0, stream>>>(A, W, bias, Co, M, K, N, scale, act, mul, mrow, addp);
  };
  auto LN256 = [&](const float* xin, const float* g_, const float* b_, float* yout){
    ln_kernel<256><<<dim3(SSQ / 8, 1, 1), blk, 0, stream>>>(xin, g_, b_, yout);
  };
  auto LN128 = [&](const float* xin, const float* g_, const float* b_, float* yout){
    ln_kernel<128><<<dim3(SSQ / 8, 1, 1), blk, 0, stream>>>(xin, g_, b_, yout);
  };

  // resize: PR0 = pair_rep @ resize.w + b ; X = PR0
  GEMM(prep, PW(67), PW(66), PR0, SSQ, 33, 256, 1.f, 0, nullptr, nullptr, nullptr);
  hipMemcpyAsync(X, PR0, BIG * sizeof(float), hipMemcpyDeviceToDevice, stream);

  // ---- triangle multiplication blocks (b0: 34=tu_in, 50=tu_out) ----
  auto TUBLK = [&](int b0, int mode){
    LN256(X, PW(b0 + 3), PW(b0 + 2), XL);                                              // e_norm
    GEMM(XL, PW(b0 + 7),  PW(b0 + 6),  B1, SSQ, 256, 128, 1.f, 0, nullptr, mask, nullptr); // le*mask
    GEMM(XL, PW(b0 + 11), PW(b0 + 10), B2, SSQ, 256, 128, 1.f, 0, nullptr, mask, nullptr); // re*mask
    GEMM(XL, PW(b0 + 9),  PW(b0 + 8),  B3, SSQ, 256, 128, 1.f, 1, B1, nullptr, nullptr);   // sig(lu)*le
    GEMM(XL, PW(b0 + 15), PW(b0 + 14), B4, SSQ, 256, 128, 1.f, 1, B2, nullptr, nullptr);   // sig(ru)*re
    GEMM(XL, PW(b0 + 5),  PW(b0 + 4),  B5, SSQ, 256, 256, 1.f, 1, nullptr, nullptr, nullptr); // sig(fu)
    trimul_kernel<<<dim3(14, 14, 16), blk, 0, stream>>>(B3, B4, B1, mode);
    LN128(B1, PW(b0 + 1), PW(b0), B2);                                                 // c_norm
    GEMM(B2, PW(b0 + 13), PW(b0 + 12), X, SSQ, 128, 256, 1.f, 0, B5, mask, X);         // rescale*fu*mask + X
  };
  TUBLK(34, 0);
  TUBLK(50, 1);

  // ---- triangle attention blocks (b0: 14=ta_in, 24=ta_out) ----
  auto TABLK = [&](int b0, int transposed){
    const float* mrow = mask;
    if (transposed) {
      transpose_kernel<<<dim3(SSQ / 256, 1, 1), blk, 0, stream>>>(X, B1, 256, 0, nullptr);
      transpose_kernel<<<dim3(SSQ / 256, 1, 1), blk, 0, stream>>>(mask, MT, 1, 0, nullptr);
      mrow = MT;
      LN256(B1, PW(b0 + 1), PW(b0), XL);
    } else {
      LN256(X, PW(b0 + 1), PW(b0), XL);
    }
    GEMM(XL, PW(b0 + 8), nullptr, B1, SSQ, 256, 256, 0.125f, 0, nullptr, nullptr, nullptr); // q * c^-0.5
    GEMM(XL, PW(b0 + 7), nullptr, B2, SSQ, 256, 256, 1.f,    0, nullptr, nullptr, nullptr); // k
    GEMM(XL, PW(b0 + 9), nullptr, B3, SSQ, 256, 256, 1.f,    0, nullptr, nullptr, nullptr); // v
    GEMM(XL, PW(b0 + 5), PW(b0 + 4), B4, SSQ, 256, 256, 1.f, 1, nullptr, nullptr, nullptr); // sig(fu)
    pb_kernel<<<dim3((SSQ * 4) / 256, 1, 1), blk, 0, stream>>>(XL, PW(b0 + 6), PBb);
    attn_kernel<<<dim3(7, 4, 224), blk, 0, stream>>>(B2, B1, B3, PBb, mrow, B4, B5);
    if (!transposed) {
      GEMM(B5, PW(b0 + 3), PW(b0 + 2), X, SSQ, 256, 256, 1.f, 0, nullptr, mask, X);    // final*mask + X
    } else {
      GEMM(B5, PW(b0 + 3), PW(b0 + 2), B1, SSQ, 256, 256, 1.f, 0, nullptr, nullptr, nullptr);
      transpose_kernel<<<dim3(SSQ / 256, 1, 1), blk, 0, stream>>>(B1, X, 256, 1, mask); // X += T(final)*mask
    }
  };
  TABLK(14, 0);
  TABLK(24, 1);

  // ---- transition MLP ----
  GEMM(X, PW(11), PW(10), TBUF, SSQ, 256, 1024, 1.f, 1, nullptr, nullptr, nullptr);     // sig(t1)
  GEMM(TBUF, PW(13), PW(12), X, SSQ, 1024, 256, 1.f, 0, nullptr, nullptr, X);           // t2 + X
  ewmadd_kernel<<<dim3(SSQ, 1, 1), blk, 0, stream>>>(X, PR0, mask);                     // *mask + PR0

  // ---- heads ----
  GEMM(X, PW(69), PW(68), B1, SSQ, 256, 256, 1.f, 1, nullptr, nullptr, nullptr);        // sig(test)
  ysum_kernel<<<dim3(224, 1, 1), blk, 0, stream>>>(B1, YS);
  head_kernel<<<dim3(224, 1, 1), blk, 0, stream>>>(YS, PW(2), PW(3), PW(4), PW(5),
                                                   PW(6), PW(7), prep, CE);
  outhead_kernel<<<dim3(SSQ / 8, 1, 1), blk, 0, stream>>>(X, PW(9), PW(8), mask, (float*)d_out);
  loss_kernel<<<dim3(1, 1, 1), blk, 0, stream>>>(CE, mask, ((float*)d_out) + SSQ);
#undef PW
}
